// SFM_Model_51187420234300
// MI455X (gfx1250) — compile-verified, matrix operand order checked by reading, not run on hardware
//
#include <hip/hip_runtime.h>
#include <math.h>

#define BSZ 256
#define TT  128
#define DF  128
#define FQ  16
#define HD  128

typedef __attribute__((ext_vector_type(16))) __bf16        v16bf;
typedef __attribute__((ext_vector_type(8)))  float         v8f;
typedef __attribute__((ext_vector_type(4)))  unsigned int  u32x4;

union ABPack { v16bf v; u32x4 u[2]; };

__device__ __forceinline__ unsigned short f2bf(float x) {
  unsigned int u = __float_as_uint(x);
  u += 0x7FFFu + ((u >> 16) & 1u);   // round-to-nearest-even
  return (unsigned short)(u >> 16);
}
__device__ __forceinline__ float hsig(float z) {
  return fminf(fmaxf(z * (1.0f / 6.0f) + 0.5f, 0.0f), 1.0f);
}

// ---- LDS layout (bytes), total 323328 < 320KB WGP limit ----
#define OFF_UB    0         // bf16 packed B-operands: 264 tiles * 32 lanes * 16 elem = 270336 B
#define OFF_XH    270336    // bf16 A staging [16 rows][256 K]  = 8192 B   (K = [x_t | h])
#define OFF_ZG    278528    // f32 gate preacts [4][16][128]    = 32768 B
#define OFF_ZF    311296    // f32 fre preacts  [16][16]        = 1024 B
#define OFF_H32   312320    // f32 h           [16][128]        = 8192 B
#define OFF_BG    320512    // f32 biases i/ste/c/o [4][128]    = 2048 B
#define OFF_BF    322560    // f32 b_fre [16]
#define OFF_BA    322624    // f32 b_a   [128]
#define OFF_UA    323136    // f32 U_a   [16]
#define OFF_CT    323200    // f32 cos table [16]
#define OFF_ST    323264    // f32 sin table [16]
#define SMEM_BYTES 323328

__global__ void __launch_bounds__(512, 1) sfm_scan_kernel(
    const float* __restrict__ input,
    const float* __restrict__ W_i,  const float* __restrict__ U_i,  const float* __restrict__ b_i,
    const float* __restrict__ W_ste,const float* __restrict__ U_ste,const float* __restrict__ b_ste,
    const float* __restrict__ W_fre,const float* __restrict__ U_fre,const float* __restrict__ b_fre,
    const float* __restrict__ W_c,  const float* __restrict__ U_c,  const float* __restrict__ b_c,
    const float* __restrict__ W_o,  const float* __restrict__ U_o,  const float* __restrict__ b_o,
    const float* __restrict__ U_a,  const float* __restrict__ b_a,
    const float* __restrict__ W_sym,const float* __restrict__ b_sym,
    float* __restrict__ out)
{
  extern __shared__ char smem[];
  unsigned short* Ub  = (unsigned short*)(smem + OFF_UB);
  unsigned short* xh  = (unsigned short*)(smem + OFF_XH);
  float* zg   = (float*)(smem + OFF_ZG);
  float* zf   = (float*)(smem + OFF_ZF);
  float* h32  = (float*)(smem + OFF_H32);
  float* bG   = (float*)(smem + OFF_BG);
  float* bF   = (float*)(smem + OFF_BF);
  float* bA   = (float*)(smem + OFF_BA);
  float* uaL  = (float*)(smem + OFF_UA);
  float* ctab = (float*)(smem + OFF_CT);
  float* stab = (float*)(smem + OFF_ST);

  const int tid  = threadIdx.x;          // 512 threads = 16 waves
  const int b0   = blockIdx.x * 16;      // 16 batch rows per workgroup
  const int w    = tid >> 5;             // wave id 0..15
  const int lane = tid & 31;
  const int mlo  = lane & 15;
  const int half = lane >> 4;

  // ------------------------------------------------------------------
  // Preload: convert fused [W;U] weights to bf16 in WMMA B-operand
  // per-lane packed layout.  Tile id: g<4 gates (8 nt x 8 kt), then fre.
  // ------------------------------------------------------------------
  const float* Wg[4] = {W_i, W_ste, W_c, W_o};
  const float* Ug[4] = {U_i, U_ste, U_c, U_o};

  for (int blk = tid; blk < 264 * 32; blk += 512) {
    int tile = blk >> 5, ln = blk & 31;
    const float *Wsrc, *Usrc;
    int ncol, nt, kt;
    if (tile < 256) {
      int g = tile >> 6, r = tile & 63;
      nt = r >> 3; kt = r & 7;
      Wsrc = Wg[g]; Usrc = Ug[g]; ncol = HD;
    } else {
      kt = tile - 256; nt = 0;
      Wsrc = W_fre; Usrc = U_fre; ncol = FQ;
    }
    int n  = nt * 16 + (ln & 15);
    int hf = ln >> 4;
    unsigned short* dst = Ub + (size_t)tile * 512 + ln * 16;
    #pragma unroll
    for (int ch = 0; ch < 2; ++ch) {
      int kb = kt * 32 + ch * 16 + hf * 8;
      #pragma unroll
      for (int e = 0; e < 8; ++e) {
        int k = kb + e;
        float v = (k < DF) ? Wsrc[k * ncol + n] : Usrc[(k - DF) * ncol + n];
        dst[ch * 8 + e] = f2bf(v);
      }
    }
  }
  if (tid < 128) {
    bG[tid]       = b_i[tid];
    bG[128 + tid] = b_ste[tid];
    bG[256 + tid] = b_c[tid];
    bG[384 + tid] = b_o[tid];
    bA[tid]       = b_a[tid];
  }
  if (tid < 16) {
    bF[tid]   = b_fre[tid];
    uaL[tid]  = U_a[tid];
    float ang = 6.2831853071795864769f * (float)tid / 16.0f;
    ctab[tid] = cosf(ang);
    stab[tid] = sinf(ang);
  }
  // h(0) = 0 in the bf16 A staging buffer
  for (int i = tid; i < 16 * 128; i += 512) {
    int m = i >> 7, h = i & 127;
    xh[m * 256 + 128 + h] = 0;
  }
  __syncthreads();

  // Per-thread oscillator state: thread owns row m_upd, 4 hidden units, all 16 freqs
  const int m_upd = tid >> 5;
  const int hb    = (tid & 31) * 4;
  float Sre[4][FQ], Sim[4][FQ];
  #pragma unroll
  for (int j = 0; j < 4; ++j)
    #pragma unroll
    for (int f = 0; f < FQ; ++f) { Sre[j][f] = 0.f; Sim[j][f] = 0.f; }

  // A registers: this wave's 16 rows x K=256, 8 K-tiles, reloaded once per step
  ABPack Areg[8];

  // One 16x16(N) x 256(K) output tile: B double-buffered from LDS,
  // A from registers, f32 accumulate, result -> z row via immediate offsets.
  // ZS is a compile-time stride so stores fold into ds_store offsets.
  auto gemm_tile = [&](int tbase, float* zdst, const int ZS) {
    v8f acc = {0.f, 0.f, 0.f, 0.f, 0.f, 0.f, 0.f, 0.f};
    const unsigned short* bb = Ub + (size_t)tbase * 512 + lane * 16;
    ABPack Bc, Bn;
    Bc.u[0] = *(const u32x4*)(bb);
    Bc.u[1] = *(const u32x4*)(bb + 8);
    #pragma unroll
    for (int kt = 0; kt < 8; ++kt) {
      if (kt < 7) {   // prefetch next B K-tile while WMMA consumes current
        Bn.u[0] = *(const u32x4*)(bb + (size_t)(kt + 1) * 512);
        Bn.u[1] = *(const u32x4*)(bb + (size_t)(kt + 1) * 512 + 8);
      }
      acc = __builtin_amdgcn_wmma_f32_16x16x32_bf16(
          false, Areg[kt].v, false, Bc.v, (short)0, acc, false, false);
      Bc = Bn;
    }
    // single live base pointer; r*ZS are compile-time ds_store offsets
    float* zrow = zdst + (half * 8) * ZS + mlo;
    #pragma unroll
    for (int r = 0; r < 8; ++r)
      zrow[r * ZS] = acc[r];
  };

  // ------------------------------------------------------------------
  // Serial scan over T steps; everything stays in LDS/registers.
  // ------------------------------------------------------------------
  #pragma unroll 1
  for (int s = 0; s < TT; ++s) {
    // stage x_t (bf16) into A buffer K=[0..127]
    {
      int m = tid >> 5, d0 = (tid & 31) * 4;
      const float* src = input + (size_t)(b0 + m) * DF * TT + s;
      #pragma unroll
      for (int j = 0; j < 4; ++j) {
        int d = d0 + j;
        xh[m * 256 + d] = f2bf(src[(size_t)d * TT]);
      }
    }
    __syncthreads();

    // Load this wave's A operand (all 8 K-tiles) into registers once
    #pragma unroll
    for (int kt = 0; kt < 8; ++kt) {
      Areg[kt].u[0] = *(const u32x4*)(xh + mlo * 256 + kt * 32 + half * 8);
      Areg[kt].u[1] = *(const u32x4*)(xh + mlo * 256 + kt * 32 + 16 + half * 8);
    }

    // Gate GEMM: z = [x_t | h] @ [W;U].  33 N-tiles over 16 waves:
    // wave w handles tiles w and w+16; wave 0 additionally the fre tile.
    {
      int g0 = w >> 3, nt0 = w & 7;
      gemm_tile(g0 * 64 + nt0 * 8, zg + (g0 * 16) * HD + nt0 * 16, HD);
      int it1 = w + 16;
      int g1 = it1 >> 3, nt1 = it1 & 7;
      gemm_tile(g1 * 64 + nt1 * 8, zg + (g1 * 16) * HD + nt1 * 16, HD);
      if (w == 0)
        gemm_tile(256, zf, FQ);
    }
    __syncthreads();

    // Elementwise recurrent update (state in registers)
    float frev[FQ];
    #pragma unroll
    for (int f = 0; f < FQ; ++f)
      frev[f] = hsig(zf[m_upd * FQ + f] + bF[f]);

    const int t = s + 1;
    float hnew[4];
    #pragma unroll
    for (int j = 0; j < 4; ++j) {
      int h = hb + j;
      float zi = zg[(0 * 16 + m_upd) * HD + h] + bG[h];
      float zs = zg[(1 * 16 + m_upd) * HD + h] + bG[128 + h];
      float zc = zg[(2 * 16 + m_upd) * HD + h] + bG[256 + h];
      float zo = zg[(3 * 16 + m_upd) * HD + h] + bG[384 + h];
      float iv = hsig(zi), sv = hsig(zs), ov = hsig(zo);
      float cv = iv * tanhf(zc);
      float accA = 0.f;
      #pragma unroll
      for (int f = 0; f < FQ; ++f) {
        int p = (t * f) & 15;                 // cos/sin(2*pi*t*f/16) is 16-periodic
        float ff = sv * frev[f];
        float re = ff * Sre[j][f] + cv * ctab[p];
        float im = ff * Sim[j][f] + cv * stab[p];
        Sre[j][f] = re; Sim[j][f] = im;
        accA += (re * re + im * im) * uaL[f];
      }
      float av = tanhf(accA + bA[h]);
      hnew[j] = ov * av;
    }
    #pragma unroll
    for (int j = 0; j < 4; ++j) {
      int h = hb + j;
      xh[m_upd * 256 + 128 + h] = f2bf(hnew[j]);
      h32[m_upd * HD + h] = hnew[j];
    }
  }
  __syncthreads();

  // Final projection h @ W_sym + b_sym
  if (tid < 16) {
    float acc = b_sym[0];
    #pragma unroll 4
    for (int h = 0; h < HD; ++h) acc += h32[tid * HD + h] * W_sym[h];
    out[b0 + tid] = acc;
  }
}

extern "C" void kernel_launch(void* const* d_in, const int* in_sizes, int n_in,
                              void* d_out, int out_size, void* d_ws, size_t ws_size,
                              hipStream_t stream) {
  (void)in_sizes; (void)n_in; (void)d_ws; (void)ws_size; (void)out_size;
  const float* input = (const float*)d_in[0];
  const float* W_i   = (const float*)d_in[1];
  const float* U_i   = (const float*)d_in[2];
  const float* b_i   = (const float*)d_in[3];
  const float* W_ste = (const float*)d_in[4];
  const float* U_ste = (const float*)d_in[5];
  const float* b_ste = (const float*)d_in[6];
  const float* W_fre = (const float*)d_in[7];
  const float* U_fre = (const float*)d_in[8];
  const float* b_fre = (const float*)d_in[9];
  const float* W_c   = (const float*)d_in[10];
  const float* U_c   = (const float*)d_in[11];
  const float* b_c   = (const float*)d_in[12];
  const float* W_o   = (const float*)d_in[13];
  const float* U_o   = (const float*)d_in[14];
  const float* b_o   = (const float*)d_in[15];
  const float* U_a   = (const float*)d_in[16];
  const float* b_a   = (const float*)d_in[17];
  const float* W_sym = (const float*)d_in[18];
  const float* b_sym = (const float*)d_in[19];
  float* out = (float*)d_out;

  hipFuncSetAttribute((const void*)sfm_scan_kernel,
                      hipFuncAttributeMaxDynamicSharedMemorySize, SMEM_BYTES);
  sfm_scan_kernel<<<BSZ / 16, 512, SMEM_BYTES, stream>>>(
      input, W_i, U_i, b_i, W_ste, U_ste, b_ste, W_fre, U_fre, b_fre,
      W_c, U_c, b_c, W_o, U_o, b_o, U_a, b_a, W_sym, b_sym, out);
}
